// CDSSM_54339926229579
// MI455X (gfx1250) — matrix operand order repaired and checked by reading, not compiled, not run on hardware
//
#include <hip/hip_runtime.h>
#include <hip/hip_bf16.h>
#include <math.h>

typedef __attribute__((ext_vector_type(16))) _Float16 v16h;
typedef __attribute__((ext_vector_type(8)))  _Float16 v8h;
typedef __attribute__((ext_vector_type(4)))  _Float16 v4h;
typedef __attribute__((ext_vector_type(8)))  float    v8f;

#define T_SEQ   2048
#define DCH     128      // word depth / channels (same for both convs)
#define KCH     128      // conv output channels
#define T1      2046     // conv1 output length
#define T2      2044     // conv2 output length
#define CHUNK   64       // conv2 outputs produced per chunk
#define XROWS   82       // x rows needed per chunk
#define H1ROWS  80       // h1 rows produced per chunk (5 tiles of 16)
#define NCHUNK  32       // ceil(2044 / 64)
#define CONV_PACK (8*3*4*32*16)   // 49152 halves = one packed conv weight (96 KB)

// Branch-free tanh using hardware transcendentals (v_exp_f32, v_rcp_f32).
// tanh(x) = 1 - 2/(exp(2x)+1). Saturates correctly: exp2->inf => 1,
// exp2->0 => -1. ~5 VALU ops vs ~25 branchy ops for libm tanhf.
__device__ __forceinline__ float tanh_fast(float x) {
    float e = __builtin_amdgcn_exp2f(x * 2.88539008177792681472f); // 2*log2(e)
    return 1.0f - 2.0f * __builtin_amdgcn_rcpf(e + 1.0f);
}

// ---------------------------------------------------------------------------
// Weight pre-pack: f32 [O=128, I=128, FL=3] -> f16 B-fragments.
// Fragment order: r = (((n_tile*3 + dk)*4 + kb)*32 + lane)*16 + e
// B layout (16x16x32 f16): lane l -> n = l&15 ; element e -> k = (e&7) + 8*(l>>4) + 16*(e>>3)
// Value: w[o=n_tile*16+n][i=kb*32+k][dk]
// ---------------------------------------------------------------------------
__global__ void cdssm_pack_w(const float* __restrict__ qw1, const float* __restrict__ qw2,
                             const float* __restrict__ dw1, const float* __restrict__ dw2,
                             _Float16* __restrict__ wpack) {
    int idx = blockIdx.x * blockDim.x + threadIdx.x;
    if (idx >= 4 * CONV_PACK) return;
    int cv = idx / CONV_PACK;
    int r  = idx % CONV_PACK;
    int e    = r & 15;       int t1 = r  >> 4;
    int lane = t1 & 31;      int t2 = t1 >> 5;
    int kb   = t2 & 3;       int t3 = t2 >> 2;
    int dk   = t3 % 3;       int n_tile = t3 / 3;
    int n = lane & 15;
    int k = (e & 7) + ((lane >> 4) << 3) + ((e >> 3) << 4);
    int i = kb * 32 + k;
    int o = n_tile * 16 + n;
    const float* w = (cv == 0) ? qw1 : (cv == 1) ? qw2 : (cv == 2) ? dw1 : dw2;
    wpack[idx] = (_Float16)w[(o * DCH + i) * 3 + dk];
}

// ---------------------------------------------------------------------------
// A-fragment load from row-major f16 LDS (stride KCH).
// A layout (16x32 f16): lane l -> m = l&15 ; elems 0..7 -> k = off..off+7,
// elems 8..15 -> k = off+16..off+23, where off = 8*(l>>4).
// Two 16-byte ds_load_b128 per lane.
// ---------------------------------------------------------------------------
__device__ __forceinline__ v16h load_a_frag(const _Float16* base, int row, int kb, int laneHi) {
    const _Float16* p = base + row * KCH + kb + (laneHi << 3);
    v8h lo = *(const v8h*)(p);
    v8h hi = *(const v8h*)(p + 16);
    v16h a;
#pragma unroll
    for (int j = 0; j < 8; ++j) { a[j] = lo[j]; a[j + 8] = hi[j]; }
    return a;
}

// ---------------------------------------------------------------------------
// Main encode: one block per sequence (256 blocks). 8 waves; wave w owns
// output channels [16w, 16w+16). conv1 -> tanh -> conv2 -> tanh -> time-max.
// ---------------------------------------------------------------------------
__global__ __launch_bounds__(256) void cdssm_encode(
    const float* __restrict__ q, const float* __restrict__ pos, const float* __restrict__ negs,
    const _Float16* __restrict__ wpack,
    const float* __restrict__ qb1, const float* __restrict__ qb2,
    const float* __restrict__ db1, const float* __restrict__ db2,
    float* __restrict__ gmax)
{
    __shared__ _Float16 xs [XROWS  * KCH];   // 21 KB
    __shared__ _Float16 h1s[H1ROWS * KCH];   // 20 KB
    __shared__ float    red[256];

    const int seq    = blockIdx.x;
    const int tid    = threadIdx.x;
    const int lane   = tid & 31;
    const int wave   = tid >> 5;
    const int laneLo = lane & 15;
    const int laneHi = lane >> 4;
    const int n0     = wave * 16;

    const float* xseq;
    const _Float16 *wp1, *wp2;
    const float *b1, *b2;
    if (seq < 64) {
        xseq = q + (size_t)seq * T_SEQ * DCH;
        wp1 = wpack;                 wp2 = wpack + CONV_PACK;
        b1 = qb1;                    b2 = qb2;
    } else {
        xseq = (seq < 128) ? (pos  + (size_t)(seq -  64) * T_SEQ * DCH)
                           : (negs + (size_t)(seq - 128) * T_SEQ * DCH);
        wp1 = wpack + 2 * CONV_PACK; wp2 = wpack + 3 * CONV_PACK;
        b1 = db1;                    b2 = db2;
    }
    // per-wave fragment region (n_tile == wave): 12 fragments of 512 halves
    const _Float16* wp1w = wp1 + (size_t)wave * 3 * 4 * 32 * 16;
    const _Float16* wp2w = wp2 + (size_t)wave * 3 * 4 * 32 * 16;

    const float bias1 = b1[n0 + laneLo];
    const float bias2 = b2[n0 + laneLo];

    float runmax = -2.0f;   // tanh outputs are in [-1, 1]

#pragma unroll 1
    for (int chunk = 0; chunk < NCHUNK; ++chunk) {
        const int t0 = chunk * CHUNK;

        // ---- Phase 1: stage x rows [t0, t0+XROWS) into LDS as f16 ----
        for (int idx = tid; idx < XROWS * 32; idx += 256) {
            int r  = idx >> 5;
            int c4 = (idx & 31) << 2;
            int g  = t0 + r;
            float4 v = make_float4(0.f, 0.f, 0.f, 0.f);
            if (g < T_SEQ) v = *(const float4*)(xseq + (size_t)g * DCH + c4);
            v4h h; h[0] = (_Float16)v.x; h[1] = (_Float16)v.y;
                   h[2] = (_Float16)v.z; h[3] = (_Float16)v.w;
            *(v4h*)(&xs[r * KCH + c4]) = h;
        }
        // speculative prefetch of next chunk's incoming rows (global_prefetch_b8)
        if (tid < CHUNK) {
            int g = t0 + XROWS + tid;
            if (g < T_SEQ) __builtin_prefetch(xseq + (size_t)g * DCH, 0, 1);
        }
        __syncthreads();

        // ---- Phase 2: conv1 (+bias) -> tanh -> h1s, 5 tiles of 16 rows ----
#pragma unroll
        for (int mt = 0; mt < H1ROWS; mt += 16) {
            v8f c;
#pragma unroll
            for (int e = 0; e < 8; ++e) c[e] = bias1;
#pragma unroll
            for (int dk = 0; dk < 3; ++dk) {
                const int row = mt + laneLo + dk;
                v16h afr[4];
#pragma unroll
                for (int kb = 0; kb < 4; ++kb)
                    afr[kb] = load_a_frag(xs, row, kb * 32, laneHi);
#pragma unroll
                for (int kb = 0; kb < 4; ++kb) {
                    v16h bfr = *(const v16h*)(wp1w + ((dk * 4 + kb) * 32 + lane) * 16);
                    c = __builtin_amdgcn_wmma_f32_16x16x32_f16(
                            false, afr[kb], false, bfr, (short)0, c, false, false);
                }
            }
#pragma unroll
            for (int e = 0; e < 8; ++e) {
                int M = e + (laneHi << 3);
                h1s[(mt + M) * KCH + n0 + laneLo] = (_Float16)tanh_fast(c[e]);
            }
        }
        __syncthreads();

        // ---- Phase 3: conv2 (+bias) -> tanh -> running max, 4 tiles ----
#pragma unroll
        for (int mt = 0; mt < CHUNK; mt += 16) {
            v8f c;
#pragma unroll
            for (int e = 0; e < 8; ++e) c[e] = bias2;
#pragma unroll
            for (int dk = 0; dk < 3; ++dk) {
                const int row = mt + laneLo + dk;
                v16h afr[4];
#pragma unroll
                for (int kb = 0; kb < 4; ++kb)
                    afr[kb] = load_a_frag(h1s, row, kb * 32, laneHi);
#pragma unroll
                for (int kb = 0; kb < 4; ++kb) {
                    v16h bfr = *(const v16h*)(wp2w + ((dk * 4 + kb) * 32 + lane) * 16);
                    c = __builtin_amdgcn_wmma_f32_16x16x32_f16(
                            false, afr[kb], false, bfr, (short)0, c, false, false);
                }
            }
#pragma unroll
            for (int e = 0; e < 8; ++e) {
                int M  = e + (laneHi << 3);
                int tg = t0 + mt + M;
                float v = tanh_fast(c[e]);
                runmax = (tg < T2) ? fmaxf(runmax, v) : runmax;
            }
        }
        __syncthreads();
    }

    // ---- reduce the two lanes covering each channel; write [seq][128] ----
    red[tid] = runmax;
    __syncthreads();
    if (tid < 128) {
        int w = tid >> 4, n = tid & 15;
        float m = fmaxf(red[w * 32 + n], red[w * 32 + n + 16]);
        gmax[(size_t)seq * KCH + w * 16 + n] = m;
    }
}

// ---------------------------------------------------------------------------
// Dense 128 -> 64 + tanh. One block per sequence, one thread per output.
// ---------------------------------------------------------------------------
__global__ void cdssm_sem(const float* __restrict__ gmax,
                          const float* __restrict__ qsw, const float* __restrict__ qsb,
                          const float* __restrict__ dsw, const float* __restrict__ dsb,
                          float* __restrict__ sem)
{
    int s = blockIdx.x;
    int l = threadIdx.x;                 // 0..63
    const float* W  = (s < 64) ? qsw : dsw;
    const float* Bv = (s < 64) ? qsb : dsb;
    float acc = Bv[l];
    const float* g = gmax + (size_t)s * KCH;
#pragma unroll 4
    for (int n = 0; n < KCH; ++n) acc += W[l * KCH + n] * g[n];
    sem[(size_t)s * 64 + l] = tanh_fast(acc);
}

// ---------------------------------------------------------------------------
// Dots + gamma + softmax. One thread per batch element.
// seq map: q=b, pos=64+b, neg_j = 128 + j*64 + b
// ---------------------------------------------------------------------------
__global__ void cdssm_out(const float* __restrict__ sem,
                          const float* __restrict__ gw, const float* __restrict__ gb,
                          float* __restrict__ out)
{
    int b = threadIdx.x;
    if (b >= 64) return;
    const float* qs = sem + (size_t)b * 64;
    float d[3];
#pragma unroll
    for (int j = 0; j < 3; ++j) {
        const float* cs = sem + (size_t)(64 + j * 64 + b) * 64;
        float acc = 0.f;
#pragma unroll 4
        for (int l = 0; l < 64; ++l) acc += qs[l] * cs[l];
        d[j] = acc;
    }
    float g = gw[0], bb = gb[0];
    float z[3], mx = -1e30f;
#pragma unroll
    for (int j = 0; j < 3; ++j) { z[j] = g * d[j] + bb; mx = fmaxf(mx, z[j]); }
    float se = 0.f;
#pragma unroll
    for (int j = 0; j < 3; ++j) { z[j] = __builtin_amdgcn_exp2f((z[j] - mx) * 1.44269504088896340736f); se += z[j]; }
    float inv = __builtin_amdgcn_rcpf(se);
#pragma unroll
    for (int j = 0; j < 3; ++j) out[b * 3 + j] = z[j] * inv;
}

// ---------------------------------------------------------------------------
extern "C" void kernel_launch(void* const* d_in, const int* in_sizes, int n_in,
                              void* d_out, int out_size, void* d_ws, size_t ws_size,
                              hipStream_t stream) {
    const float* q    = (const float*)d_in[0];
    const float* pos  = (const float*)d_in[1];
    const float* negs = (const float*)d_in[2];
    const float* qw1  = (const float*)d_in[3];
    const float* qb1  = (const float*)d_in[4];
    const float* qw2  = (const float*)d_in[5];
    const float* qb2  = (const float*)d_in[6];
    const float* qsw  = (const float*)d_in[7];
    const float* qsb  = (const float*)d_in[8];
    const float* dw1  = (const float*)d_in[9];
    const float* db1  = (const float*)d_in[10];
    const float* dw2  = (const float*)d_in[11];
    const float* db2  = (const float*)d_in[12];
    const float* dsw  = (const float*)d_in[13];
    const float* dsb  = (const float*)d_in[14];
    const float* gw   = (const float*)d_in[15];
    const float* gb   = (const float*)d_in[16];
    (void)in_sizes; (void)n_in; (void)out_size; (void)ws_size;

    char* ws = (char*)d_ws;
    _Float16* wpack = (_Float16*)ws;                                   // 384 KB
    float* gmax = (float*)(ws + 4 * CONV_PACK * sizeof(_Float16));     // 128 KB
    float* sem  = (float*)(ws + 4 * CONV_PACK * sizeof(_Float16)
                              + 256 * KCH * sizeof(float));            //  64 KB

    cdssm_pack_w<<<(4 * CONV_PACK + 255) / 256, 256, 0, stream>>>(qw1, qw2, dw1, dw2, wpack);
    cdssm_encode<<<256, 256, 0, stream>>>(q, pos, negs, wpack, qb1, qb2, db1, db2, gmax);
    cdssm_sem<<<256, 64, 0, stream>>>(gmax, qsw, qsb, dsw, dsb, sem);
    cdssm_out<<<1, 64, 0, stream>>>(sem, gw, gb, (float*)d_out);
}
